// BaseConch_16406775071373
// MI455X (gfx1250) — matrix operand order repaired and checked by reading, not compiled, not run on hardware
//
#include <hip/hip_runtime.h>
#include <hip/hip_bf16.h>

// ---------------------------------------------------------------------------
// MI455X (gfx1250) implementation.
//
// Precision: all intermediate node/edge features kept in bf16 (edge table
// 400000x128 bf16 = 102 MB -> resident in the 192 MB L2, so random s=16
// gathers are L2 hits). All matrix math is V_WMMA_F32_16X16X32_BF16 with f32
// accumulators; softmax / ELU / final output in f32.
//
// Tiling: one wave owns a 16-query tile. Gathered 256 B neighbor rows are
// loaded by global_load_b128 directly in WMMA A-fragment layout (no LDS
// staging of activations). Pre-swizzled q/k/v weight fragments are staged
// once per workgroup into LDS (96 KB of the 320 KB WGP pool) so every WMMA
// B-operand is a ds_load_b128 pair instead of a global load -> the matrix
// pipe no longer serializes on VMEM, and VMEM bandwidth is left for gathers.
// Softmax lives entirely in registers via wave32 shfl butterflies.
// ---------------------------------------------------------------------------

typedef __bf16 bf16x8_t  __attribute__((ext_vector_type(8)));
typedef __bf16 bf16x16_t __attribute__((ext_vector_type(16)));
typedef float  f32x8_t   __attribute__((ext_vector_type(8)));

union FragU { bf16x16_t v; bf16x8_t h[2]; };

#define WMMA_BF16(a, b, c) \
  __builtin_amdgcn_wmma_f32_16x16x32_bf16(false, (a), false, (b), (short)0, (c), false, false)

// A-fragment (16x32 bf16, M = query row) from a row-major bf16 row pointer.
// Lane L (<16): K 0-7 in v0-3, K 16-23 in v4-7. Lane L (>=16): K 8-15 / 24-31.
__device__ __forceinline__ bf16x16_t load_a_frag(const __bf16* __restrict__ row,
                                                 int t, int half) {
  FragU f;
  f.h[0] = *(const bf16x8_t*)(row + t * 32 + half * 8);
  f.h[1] = *(const bf16x8_t*)(row + t * 32 + 16 + half * 8);
  return f.v;
}

// B-fragment (32x16 bf16, N = output col): frag fidx stored as
// [32 lanes][16 bf16] contiguous -> 32 B per lane (2x b128).
__device__ __forceinline__ bf16x16_t load_b_frag(const __bf16* w, int fidx,
                                                 int lane) {
  const bf16x8_t* p = (const bf16x8_t*)(w + ((size_t)(fidx * 32 + lane)) * 16);
  FragU f;
  f.h[0] = p[0];
  f.h[1] = p[1];
  return f.v;
}

// ---------------------------------------------------------------------------
// f32 -> bf16 row-major convert
// ---------------------------------------------------------------------------
__global__ void cvt_kernel(const float* __restrict__ in, __bf16* __restrict__ out,
                           long long n) {
  long long i = (long long)blockIdx.x * blockDim.x + threadIdx.x;
  if (i < n) out[i] = (__bf16)in[i];
}

// ---------------------------------------------------------------------------
// Weight swizzle into B-fragment order.
// mode 0: W is [Din, 128] row-major f32 (Wprep / Wedgeprep)
// mode 1: W is [H=4, D=128, K=32] f32 (attention weights); logical col n=h*32+k
// Output frag f = t*8+nt : element (f, lane, e) =
//   W[d = t*32+(e>>3)*16+(lane>>4)*8+(e&7)][n = nt*16+(lane&15)]
// ---------------------------------------------------------------------------
__global__ void swizzle_kernel(const float* __restrict__ W, __bf16* __restrict__ out,
                               int mode, int nfrag) {
  int tid = blockIdx.x * blockDim.x + threadIdx.x;
  if (tid >= nfrag * 512) return;
  int e = tid & 15;
  int ln = (tid >> 4) & 31;
  int f = tid >> 9;
  int t = f >> 3, nt = f & 7;
  int d = t * 32 + (e >> 3) * 16 + (ln >> 4) * 8 + (e & 7);
  int n = nt * 16 + (ln & 15);
  float v;
  if (mode == 0) v = W[(size_t)d * 128 + n];
  else           v = W[(size_t)(n >> 5) * 4096 + (size_t)d * 32 + (n & 31)];
  out[((size_t)(f * 32 + ln)) * 16 + e] = (__bf16)v;
}

// ---------------------------------------------------------------------------
// WMMA GEMM: out_bf16[rows,128] = A_bf16[rows, KS*32] @ Wswz
// one wave per 16-row tile, 8 waves per workgroup, W staged in LDS.
// ---------------------------------------------------------------------------
template <int KS>
__global__ __launch_bounds__(256) void gemm_kernel(const __bf16* __restrict__ A,
                                                   const __bf16* __restrict__ W,
                                                   __bf16* __restrict__ out,
                                                   int rows) {
  __shared__ __align__(16) __bf16 s_w[16384];  // up to 32 frags (32 KB)
  {
    uint4* dst = (uint4*)s_w;
    const uint4* src = (const uint4*)W;
#pragma unroll
    for (int i = threadIdx.x; i < KS * 1024; i += 256) dst[i] = src[i];
  }
  __syncthreads();

  const int wid = blockIdx.x * 8 + (threadIdx.x >> 5);
  const int rbase = wid * 16;
  if (rbase >= rows) return;
  const int lane = threadIdx.x & 31, half = lane >> 4, mrow = lane & 15;

  bf16x16_t a[KS];
  const __bf16* ar = A + (size_t)(rbase + mrow) * (KS * 32);
#pragma unroll
  for (int t = 0; t < KS; ++t) a[t] = load_a_frag(ar, t, half);

#pragma unroll
  for (int nt = 0; nt < 8; ++nt) {
    f32x8_t acc = {0.f, 0.f, 0.f, 0.f, 0.f, 0.f, 0.f, 0.f};
#pragma unroll
    for (int t = 0; t < KS; ++t)
      acc = WMMA_BF16(a[t], load_b_frag(s_w, t * 8 + nt, lane), acc);
#pragma unroll
    for (int j = 0; j < 8; ++j) {
      const int m = j + half * 8, col = nt * 16 + mrow;
      out[(size_t)(rbase + m) * 128 + col] = (__bf16)acc[j];
    }
  }
}

// ---------------------------------------------------------------------------
// Fused multi-head attention aggregation (attn_agg in the reference).
//   x    : [nq,128] bf16   (query source)
//   src  : [*,128]  bf16   (neighbor table, gathered through idx)
//   idx  : [nq,S]   int
//   wqkv : 3 contiguous pre-swizzled weight matrices (q,k,v), 96 KB total
//   fout : optional f32 output slice (stride ldo); bout: bf16 next-features
// One wave per 16-query tile; H=4 heads x K=32; softmax fully in registers.
// ---------------------------------------------------------------------------
template <int S>
__global__ __launch_bounds__(256) void attn_kernel(
    const __bf16* __restrict__ x, const __bf16* __restrict__ src,
    const int* __restrict__ idx, const __bf16* __restrict__ wqkv,
    float* __restrict__ fout, int ldo,
    __bf16* __restrict__ bout, int nq) {
  extern __shared__ __align__(16) char smem[];
  __bf16* sw_q = (__bf16*)smem;            // 16384 elems each
  __bf16* sw_k = sw_q + 16384;
  __bf16* sw_v = sw_k + 16384;

  // Cooperative stage of all three weight matrices (6144 x 16B), then a
  // workgroup barrier BEFORE the wave-uniform early exit (barrier-safe).
  {
    uint4* dst = (uint4*)smem;
    const uint4* s0 = (const uint4*)wqkv;
#pragma unroll
    for (int i = threadIdx.x; i < 6144; i += 256) dst[i] = s0[i];
  }
  __syncthreads();

  const int wid = blockIdx.x * 8 + (threadIdx.x >> 5);
  const int qbase = wid * 16;
  if (qbase >= nq) return;  // wave-uniform; EXEC stays all-ones for WMMA
  const int lane = threadIdx.x & 31, half = lane >> 4, mrow = lane & 15;

  // Gather indices for this tile (row = this lane's query).
  int nid[S];
#pragma unroll
  for (int s = 0; s < S; ++s) nid[s] = idx[(size_t)(qbase + mrow) * S + s];

  // X tile A-fragments.
  bf16x16_t xa[4];
  {
    const __bf16* xr = x + (size_t)(qbase + mrow) * 128;
#pragma unroll
    for (int t = 0; t < 4; ++t) xa[t] = load_a_frag(xr, t, half);
  }

  // Q = X @ Wq  -> 8 C-tiles [16 x 128] f32
  f32x8_t q[8];
#pragma unroll
  for (int nt = 0; nt < 8; ++nt) {
    f32x8_t acc = {0.f, 0.f, 0.f, 0.f, 0.f, 0.f, 0.f, 0.f};
#pragma unroll
    for (int t = 0; t < 4; ++t)
      acc = WMMA_BF16(xa[t], load_b_frag(sw_q, t * 8 + nt, lane), acc);
    q[nt] = acc;
  }

  // Scores. After the 16-lane butterfly every lane of a half holds
  // score[m = j + half*8, h, s]; the lane with mrow==s keeps it in areg.
  float areg[8][4];
#pragma unroll
  for (int j = 0; j < 8; ++j)
#pragma unroll
    for (int h = 0; h < 4; ++h) areg[j][h] = 0.f;

#pragma unroll
  for (int s = 0; s < S; ++s) {
    const __bf16* nr = src + (size_t)nid[s] * 128;
    if (s + 1 < S) __builtin_prefetch(src + (size_t)nid[s + 1] * 128, 0, 1);
    bf16x16_t na[4];
#pragma unroll
    for (int t = 0; t < 4; ++t) na[t] = load_a_frag(nr, t, half);

#pragma unroll
    for (int h = 0; h < 4; ++h) {
      float ps[8];
#pragma unroll
      for (int j = 0; j < 8; ++j) ps[j] = 0.f;
#pragma unroll
      for (int u = 0; u < 2; ++u) {
        const int nt = 2 * h + u;
        f32x8_t kk = {0.f, 0.f, 0.f, 0.f, 0.f, 0.f, 0.f, 0.f};
#pragma unroll
        for (int t = 0; t < 4; ++t)
          kk = WMMA_BF16(na[t], load_b_frag(sw_k, t * 8 + nt, lane), kk);
#pragma unroll
        for (int j = 0; j < 8; ++j) ps[j] += q[nt][j] * kk[j];
      }
#pragma unroll
      for (int j = 0; j < 8; ++j) {
        float r = ps[j];
        r += __shfl_xor(r, 1, 32);
        r += __shfl_xor(r, 2, 32);
        r += __shfl_xor(r, 4, 32);
        r += __shfl_xor(r, 8, 32);
        if (mrow == s) areg[j][h] = r;
      }
    }
  }

  // Register softmax over s: butterfly across the S score-holding lanes
  // (S is a power of two; lanes mrow>=S carry ignored garbage).
  const float scale = 0.17677669529663687f;  // 1/sqrt(32)
#pragma unroll
  for (int j = 0; j < 8; ++j)
#pragma unroll
    for (int h = 0; h < 4; ++h) {
      float v = areg[j][h];
      float mx = v;
#pragma unroll
      for (int mask = 1; mask < S; mask <<= 1)
        mx = fmaxf(mx, __shfl_xor(mx, mask, 32));
      float e = __expf((v - mx) * scale);
      float z = e;
#pragma unroll
      for (int mask = 1; mask < S; mask <<= 1) z += __shfl_xor(z, mask, 32);
      areg[j][h] = e / z;
    }

  // out = attn @ (neigh @ Wv): attention weight broadcast from lane half*16+s.
  f32x8_t o[8];
#pragma unroll
  for (int nt = 0; nt < 8; ++nt) {
    f32x8_t z0 = {0.f, 0.f, 0.f, 0.f, 0.f, 0.f, 0.f, 0.f};
    o[nt] = z0;
  }
#pragma unroll
  for (int s = 0; s < S; ++s) {
    const __bf16* nr = src + (size_t)nid[s] * 128;
    bf16x16_t na[4];
#pragma unroll
    for (int t = 0; t < 4; ++t) na[t] = load_a_frag(nr, t, half);
#pragma unroll
    for (int h = 0; h < 4; ++h) {
      float aw[8];
#pragma unroll
      for (int j = 0; j < 8; ++j)
        aw[j] = __shfl(areg[j][h], (half << 4) + s, 32);
#pragma unroll
      for (int u = 0; u < 2; ++u) {
        const int nt = 2 * h + u;
        f32x8_t vv = {0.f, 0.f, 0.f, 0.f, 0.f, 0.f, 0.f, 0.f};
#pragma unroll
        for (int t = 0; t < 4; ++t)
          vv = WMMA_BF16(na[t], load_b_frag(sw_v, t * 8 + nt, lane), vv);
#pragma unroll
        for (int j = 0; j < 8; ++j) o[nt][j] += aw[j] * vv[j];
      }
    }
  }

  // ELU + stores (f32 output slice and bf16 next-features).
#pragma unroll
  for (int nt = 0; nt < 8; ++nt) {
#pragma unroll
    for (int j = 0; j < 8; ++j) {
      float v = o[nt][j];
      v = v > 0.f ? v : (__expf(v) - 1.f);
      const int m = j + half * 8, col = nt * 16 + mrow;
      if (fout) fout[(size_t)(qbase + m) * ldo + col] = v;
      bout[(size_t)(qbase + m) * 128 + col] = (__bf16)v;
    }
  }
}

// ---------------------------------------------------------------------------
// Host orchestration
// ---------------------------------------------------------------------------
extern "C" void kernel_launch(void* const* d_in, const int* in_sizes, int n_in,
                              void* d_out, int out_size, void* d_ws, size_t ws_size,
                              hipStream_t stream) {
  (void)in_sizes; (void)n_in; (void)out_size; (void)ws_size;
  constexpr int N = 50000, Sn = 16, E = 400000, NMP = 2, DEPTH = 2;
  constexpr size_t MATE = 16384;        // elements per swizzled 128x128 matrix
  constexpr unsigned SMEM_ATTN = 3 * MATE * 2;  // 96 KB dynamic LDS

  const float* feats     = (const float*)d_in[0];
  const float* node_emb  = (const float*)d_in[1];
  const float* Wprep     = (const float*)d_in[2];
  const float* edge_emb  = (const float*)d_in[3];
  const float* Wedgeprep = (const float*)d_in[4];
  const float* Wfam[6]   = {(const float*)d_in[5], (const float*)d_in[6],
                            (const float*)d_in[7], (const float*)d_in[8],
                            (const float*)d_in[9], (const float*)d_in[10]};
  const int* n2e = (const int*)d_in[11];
  const int* adj = (const int*)d_in[12];
  float* out = (float*)d_out;

  // Workspace carve-out.
  char* ws = (char*)d_ws;
  size_t off = 0;
  auto carve = [&](size_t bytes) {
    size_t o = off;
    off = (off + bytes + 255) & ~(size_t)255;
    return o;
  };
  __bf16* featsb  = (__bf16*)(ws + carve((size_t)N * 128 * 2));
  __bf16* nembb   = (__bf16*)(ws + carve((size_t)N * 128 * 2));
  __bf16* wprep_s = (__bf16*)(ws + carve(32 * 512 * 2));
  __bf16* wedge_s = (__bf16*)(ws + carve((size_t)NMP * 16 * 512 * 2));
  __bf16* wattn   = (__bf16*)(ws + carve((size_t)24 * MATE * 2));
  __bf16* eembb   = (__bf16*)(ws + carve((size_t)E * 64 * 2));
  __bf16* featA   = (__bf16*)(ws + carve((size_t)N * 128 * 2));
  __bf16* featB   = (__bf16*)(ws + carve((size_t)N * 128 * 2));
  __bf16* edgeA   = (__bf16*)(ws + carve((size_t)E * 128 * 2));
  __bf16* edgeB   = (__bf16*)(ws + carve((size_t)E * 128 * 2));

  auto cvt = [&](const float* s, __bf16* d, long long n) {
    cvt_kernel<<<(unsigned)((n + 255) / 256), 256, 0, stream>>>(s, d, n);
  };
  auto swz = [&](const float* s, __bf16* d, int mode, int nfrag) {
    int nt = nfrag * 512;
    swizzle_kernel<<<(nt + 255) / 256, 256, 0, stream>>>(s, d, mode, nfrag);
  };

  // One-time (per launch) converts + weight swizzles.
  // wattn layout: [(kind e/n), mp, l] -> 3 contiguous matrices (q, k, v).
  cvt(feats, featsb, (long long)N * 128);
  cvt(node_emb, nembb, (long long)N * 128);
  swz(Wprep, wprep_s, 0, 32);
  for (int mp = 0; mp < NMP; ++mp)
    swz(Wedgeprep + (size_t)mp * 64 * 128, wedge_s + (size_t)mp * 16 * 512, 0, 16);
  for (int f = 0; f < 6; ++f) {
    const int kind = (f < 3) ? 0 : 1, which = f % 3;
    for (int mp = 0; mp < NMP; ++mp)
      for (int l = 0; l < DEPTH; ++l) {
        size_t midx = (size_t)(((kind * NMP + mp) * DEPTH + l) * 3 + which);
        swz(Wfam[f] + (size_t)(mp * DEPTH + l) * MATE, wattn + midx * MATE, 1, 32);
      }
  }

  const int tilesN = N / 16, tilesE = E / 16;  // both exact
  const int gN = (tilesN + 7) / 8, gE = (tilesE + 7) / 8;

  for (int mp = 0; mp < NMP; ++mp) {
    // Prep GEMMs.
    gemm_kernel<4><<<gN, 256, 0, stream>>>(featsb, wprep_s, featA, N);
    cvt(edge_emb + (size_t)mp * E * 64, eembb, (long long)E * 64);
    gemm_kernel<2><<<gE, 256, 0, stream>>>(eembb, wedge_s + (size_t)mp * 16 * 512,
                                           edgeA, E);

    __bf16 *fc = featA, *fn = featB, *ec = edgeA, *en = edgeB;
    for (int l = 0; l < DEPTH; ++l) {
      const __bf16* w_e = wattn + (size_t)(((0 * NMP + mp) * DEPTH + l) * 3) * MATE;
      const __bf16* w_n = wattn + (size_t)(((1 * NMP + mp) * DEPTH + l) * 3) * MATE;

      // Edge update: x = current edges, neighbors = 2 endpoint node feats.
      attn_kernel<2><<<gE, 256, SMEM_ATTN, stream>>>(
          ec, fc, adj + (size_t)mp * E * 2, w_e, nullptr, 0, en, E);
      // Node update: x = node_emb (l==0) or current feats; neighbors = OLD edges.
      const __bf16* xp = (l == 0) ? nembb : fc;
      attn_kernel<16><<<gN, 256, SMEM_ATTN, stream>>>(
          xp, ec, n2e + (size_t)mp * N * Sn, w_n,
          out + (size_t)mp * N * 256 + (size_t)l * 128, 256, fn, N);
      __bf16* t;
      t = fc; fc = fn; fn = t;
      t = ec; ec = en; en = t;
    }
  }
}
